// QEPolaLinearAttention_17557826306292
// MI455X (gfx1250) — matrix operand order repaired and checked by reading, not compiled, not run on hardware
//
#include <hip/hip_runtime.h>
#include <hip/hip_bf16.h>
#include <math.h>

// ---------------- problem constants ----------------
#define BATCH   8
#define SEQ_N   3136     // 56*56
#define CH      512
#define NHEADS  8
#define HDIM    64       // CH / NHEADS
#define IMG     56
#define BH      (BATCH*NHEADS)   // 64

typedef __attribute__((ext_vector_type(16))) __bf16 v16bf;
typedef __attribute__((ext_vector_type(8)))  float  v8f;

// ---- CDNA5 async global->LDS copy (ASYNCcnt-tracked, bypasses VGPRs) ----
__device__ __forceinline__ void async_copy_b128(unsigned lds_off, const void* gaddr) {
  asm volatile("global_load_async_to_lds_b128 %0, %1, off"
               :: "v"(lds_off), "v"((unsigned long long)(uintptr_t)gaddr)
               : "memory");
}
__device__ __forceinline__ void wait_async0() {
  asm volatile("s_wait_asynccnt 0x0" ::: "memory");
}

// ---------------- WMMA GEMM: C[M,Nout] = alpha * A[M,K] x B + bias ----------------
// A: bf16 row-major [M,K] (lda)
// B: bf16, either [Nout,K] (b_row_major=0, W-transposed einsum layout) or [K,Nout] (b_row_major=1)
// C: f32 row-major [M,Nout] (ldc).  Batched via blockIdx.z with per-operand strides.
#define BM 128
#define BN 64
#define BK 64

__global__ __launch_bounds__(256)
void wmma_gemm_bf16(const __bf16* __restrict__ A, int lda, long long strideA,
                    const __bf16* __restrict__ Bm, int ldb, long long strideB,
                    float* __restrict__ C, int ldc, long long strideC,
                    int M, int Nout, int K, float alpha,
                    const float* __restrict__ bias, int b_row_major)
{
  __shared__ __align__(16) __bf16 As[2][BM][BK];   // 32 KB
  __shared__ __align__(16) __bf16 Bs[2][BN][BK];   // 16 KB, stored [n][k]

  const int batch = blockIdx.z;
  A  += (long long)batch * strideA;
  Bm += (long long)batch * strideB;
  C  += (long long)batch * strideC;

  const int m0  = blockIdx.y * BM;
  const int n0  = blockIdx.x * BN;
  const int tid = threadIdx.x;
  const int wave  = tid >> 5;
  const int lane  = tid & 31;
  const int wm    = (wave & 3) * 32;     // 4 waves along M
  const int wn    = (wave >> 2) * 32;    // 2 waves along N
  const int lhalf = lane >> 4;
  const int lrow  = lane & 15;

  v8f acc[2][2] = {};

  // ---- async tile staging (double-buffered) ----
  auto stageA = [&](int buf, int k0) {
    // 128x64 bf16 tile: 256 thr x 32 elems (4 x b128 async per thread)
    const int row = tid >> 1;
    const int cb  = (tid & 1) << 5;      // 0 or 32
    if (m0 + row < M) {
      const __bf16* src = A + (long long)(m0 + row) * lda + k0 + cb;
      unsigned l = (unsigned)(uintptr_t)&As[buf][row][cb];
      async_copy_b128(l,      src);
      async_copy_b128(l + 16, src + 8);
      async_copy_b128(l + 32, src + 16);
      async_copy_b128(l + 48, src + 24);
    }
  };
  auto stageB = [&](int buf, int k0) {
    if (!b_row_major) {
      // 64x64 tile, [n][k] source order == LDS order: 2 x b128 async per thread
      const int row = tid >> 2;
      const int cb  = (tid & 3) << 4;    // 0,16,32,48
      if (n0 + row < Nout) {
        const __bf16* src = Bm + (long long)(n0 + row) * ldb + k0 + cb;
        unsigned l = (unsigned)(uintptr_t)&Bs[buf][row][cb];
        async_copy_b128(l,      src);
        async_copy_b128(l + 16, src + 8);
      }
    } else {
      // B is [K,Nout]: coalesced b128 read along N, transpose-scatter into [n][k]
      const int k  = tid & 63;
      const int bn = (tid >> 6) << 4;    // 0,16,32,48
      #pragma unroll
      for (int half = 0; half < 2; ++half) {
        const int nb = bn + half * 8;
        __bf16 tmp[8];
        if (n0 + nb < Nout) {
          *(uint4*)tmp = *(const uint4*)(Bm + (long long)(k0 + k) * ldb + n0 + nb);
        } else {
          #pragma unroll
          for (int j = 0; j < 8; ++j) tmp[j] = (__bf16)0.0f;
        }
        #pragma unroll
        for (int j = 0; j < 8; ++j) Bs[buf][nb + j][k] = tmp[j];
      }
    }
  };

  stageA(0, 0);
  stageB(0, 0);

  int buf = 0;
  for (int k0 = 0; k0 < K; k0 += BK) {
    wait_async0();        // our async writes into As/Bs[buf] are done
    __syncthreads();      // everyone's are done; prior reads of buf^1 finished
    if (k0 + BK < K) {    // overlap next-tile DMA with this tile's WMMAs
      stageA(buf ^ 1, k0 + BK);
      stageB(buf ^ 1, k0 + BK);
    }

    #pragma unroll
    for (int kk = 0; kk < BK; kk += 32) {
      // ---- fragments per ISA 7.12.2 wave32 layouts ----
      v16bf af[2], bfrag[2];
      #pragma unroll
      for (int a = 0; a < 2; ++a) {
        const __bf16* ap = &As[buf][wm + a * 16 + lrow][kk + lhalf * 8];
        ((uint4*)&af[a])[0] = *(const uint4*)ap;          // K = k0f .. k0f+7
        ((uint4*)&af[a])[1] = *(const uint4*)(ap + 16);   // K = k0f+16 .. k0f+23
      }
      #pragma unroll
      for (int b = 0; b < 2; ++b) {
        const __bf16* bp = &Bs[buf][wn + b * 16 + lrow][kk + lhalf * 8];
        ((uint4*)&bfrag[b])[0] = *(const uint4*)bp;       // K = k0f .. k0f+7
        ((uint4*)&bfrag[b])[1] = *(const uint4*)(bp + 16);   // K = k0f+16 .. k0f+23
      }
      #pragma unroll
      for (int a = 0; a < 2; ++a)
        #pragma unroll
        for (int b = 0; b < 2; ++b)
          acc[a][b] = __builtin_amdgcn_wmma_f32_16x16x32_bf16(
              false, af[a], false, bfrag[b], (short)0, acc[a][b], false, false);
    }
    buf ^= 1;
  }

  // ---- store: VGPR i -> (M = base + i + 8*lhalf, N = base + lrow) ----
  #pragma unroll
  for (int a = 0; a < 2; ++a)
    #pragma unroll
    for (int b = 0; b < 2; ++b) {
      const int col = n0 + wn + b * 16 + lrow;
      if (col >= Nout) continue;
      const float bv = bias ? bias[col] : 0.0f;
      #pragma unroll
      for (int i = 0; i < 8; ++i) {
        const int row = m0 + wm + a * 16 + lhalf * 8 + i;
        if (row < M)
          C[(long long)row * ldc + col] = alpha * acc[a][b][i] + bv;
      }
    }
}

// ---------------- elementwise / reduction kernels (all f32 math) ----------------

__global__ void cast_f32_bf16(const float* __restrict__ src, __bf16* __restrict__ dst, int n) {
  int i = blockIdx.x * blockDim.x + threadIdx.x;
  if (i < n) dst[i] = (__bf16)src[i];
}

// 3x3 depthwise enhance conv; q read from qg[:, :, :C]; qe layout [B][C][N]
__global__ void enhance_conv3(const float* __restrict__ qg, const float* __restrict__ enh_w,
                              float* __restrict__ qe) {
  int i = blockIdx.x * blockDim.x + threadIdx.x;
  const int total = BATCH * CH * SEQ_N;
  if (i >= total) return;
  int n = i % SEQ_N, c = (i / SEQ_N) % CH, b = i / (SEQ_N * CH);
  int y = n / IMG, x = n % IMG;
  float s = 0.0f;
  #pragma unroll
  for (int ky = 0; ky < 3; ++ky)
    #pragma unroll
    for (int kx = 0; kx < 3; ++kx) {
      int yy = y + ky - 1, xx = x + kx - 1;
      if (yy >= 0 && yy < IMG && xx >= 0 && xx < IMG)
        s += enh_w[c * 9 + ky * 3 + kx] *
             qg[((long long)b * SEQ_N + yy * IMG + xx) * (2 * CH) + c];
    }
  qe[i] = s;
}

__global__ __launch_bounds__(256)
void bn_stats(const float* __restrict__ qe, float* __restrict__ mean, float* __restrict__ rstd) {
  const int c = blockIdx.x;
  float s = 0.0f, s2 = 0.0f;
  for (int idx = threadIdx.x; idx < BATCH * SEQ_N; idx += 256) {
    int b = idx / SEQ_N, n = idx % SEQ_N;
    float v = qe[((long long)b * CH + c) * SEQ_N + n];
    s += v; s2 += v * v;
  }
  __shared__ float sh[256], sh2[256];
  sh[threadIdx.x] = s; sh2[threadIdx.x] = s2;
  __syncthreads();
  for (int o = 128; o > 0; o >>= 1) {
    if (threadIdx.x < o) { sh[threadIdx.x] += sh[threadIdx.x + o]; sh2[threadIdx.x] += sh2[threadIdx.x + o]; }
    __syncthreads();
  }
  if (threadIdx.x == 0) {
    const float inv = 1.0f / (float)(BATCH * SEQ_N);
    float m = sh[0] * inv;
    float var = sh2[0] * inv - m * m;
    mean[c] = m;
    rstd[c] = rsqrtf(var + 1e-5f);
  }
}

// q = (q + ew*relu(BN(qe))) / softplus(scale); k = (k + pos) / softplus(scale)   (in place)
__global__ void fuse_qk(float* __restrict__ qg, float* __restrict__ kv,
                        const float* __restrict__ qe,
                        const float* __restrict__ mean, const float* __restrict__ rstd,
                        const float* __restrict__ gamma, const float* __restrict__ beta,
                        const float* __restrict__ scale_p, const float* __restrict__ ew_p,
                        const float* __restrict__ pos_enc) {
  int i = blockIdx.x * blockDim.x + threadIdx.x;
  const int total = BATCH * SEQ_N * CH;
  if (i >= total) return;
  int c = i % CH, n = (i / CH) % SEQ_N, b = i / (CH * SEQ_N);
  float inv_scale = 1.0f / log1pf(expf(scale_p[c]));
  long long idx = ((long long)b * SEQ_N + n) * (2 * CH) + c;
  float e = (qe[((long long)b * CH + c) * SEQ_N + n] - mean[c]) * rstd[c] * gamma[c] + beta[c];
  e = fmaxf(e, 0.0f);
  qg[idx] = (qg[idx] + ew_p[0] * e) * inv_scale;
  kv[idx] = (kv[idx] + pos_enc[(long long)n * CH + c]) * inv_scale;
}

// polarity features in WMMA-friendly bf16 layouts
__global__ void features(const float* __restrict__ qg, const float* __restrict__ kv,
                         const float* __restrict__ power_p,
                         __bf16* __restrict__ QS, __bf16* __restrict__ QO,
                         __bf16* __restrict__ KFT, __bf16* __restrict__ VB) {
  int i = blockIdx.x * blockDim.x + threadIdx.x;
  const int total = BATCH * NHEADS * SEQ_N * HDIM;
  if (i >= total) return;
  int e = i % HDIM, n = (i / HDIM) % SEQ_N, h = (i / (HDIM * SEQ_N)) % NHEADS,
      b = i / (HDIM * SEQ_N * NHEADS);
  int c = h * HDIM + e, bh = b * NHEADS + h;
  float p = 1.0f + 4.0f / (1.0f + expf(-power_p[c]));
  long long base = ((long long)b * SEQ_N + n) * (2 * CH);
  float q = qg[base + c], k = kv[base + c], v = kv[base + CH + c];
  float qp = powf(fmaxf(q, 0.0f), p), qn = powf(fmaxf(-q, 0.0f), p);
  float kp = powf(fmaxf(k, 0.0f), p), kn = powf(fmaxf(-k, 0.0f), p);
  long long qsi = ((long long)bh * SEQ_N + n) * 128 + e;
  QS[qsi] = (__bf16)qp; QS[qsi + HDIM] = (__bf16)qn;        // q_sim = [q+, q-]
  QO[qsi] = (__bf16)qn; QO[qsi + HDIM] = (__bf16)qp;        // q_opp = [q-, q+]
  long long kfi = ((long long)bh * 128 + e) * SEQ_N + n;    // kf stored transposed [bh][2d][n]
  KFT[kfi] = (__bf16)kp; KFT[kfi + (long long)HDIM * SEQ_N] = (__bf16)kn;
  VB[((long long)bh * SEQ_N + n) * HDIM + e] = (__bf16)v;
}

__global__ __launch_bounds__(256)
void kmean_kernel(const __bf16* __restrict__ KFT, float* __restrict__ kmean) {
  const int row = blockIdx.x;   // BH*128 rows
  float s = 0.0f;
  for (int n = threadIdx.x; n < SEQ_N; n += 256)
    s += (float)KFT[(long long)row * SEQ_N + n];
  __shared__ float sh[256];
  sh[threadIdx.x] = s;
  __syncthreads();
  for (int o = 128; o > 0; o >>= 1) {
    if (threadIdx.x < o) sh[threadIdx.x] += sh[threadIdx.x + o];
    __syncthreads();
  }
  if (threadIdx.x == 0) kmean[row] = sh[0] / (float)SEQ_N;
}

__global__ void zfactors(const __bf16* __restrict__ QS, const __bf16* __restrict__ QO,
                         const float* __restrict__ kmean,
                         float* __restrict__ Z, float* __restrict__ Z2) {
  int i = blockIdx.x * blockDim.x + threadIdx.x;
  const int total = BH * SEQ_N;
  if (i >= total) return;
  int n = i % SEQ_N, bh = i / SEQ_N;
  const __bf16* qs = QS + ((long long)bh * SEQ_N + n) * 128;
  const __bf16* qo = QO + ((long long)bh * SEQ_N + n) * 128;
  const float* km = kmean + bh * 128;
  float s1 = 0.0f, s2 = 0.0f;
  #pragma unroll 8
  for (int d2 = 0; d2 < 128; ++d2) {
    float k = km[d2];
    s1 += (float)qs[d2] * k;
    s2 += (float)qo[d2] * k;
  }
  Z[i]  = 1.0f / (s1 + 1e-6f);
  Z2[i] = 1.0f / (s2 + 1e-6f);
}

// 5x5 depthwise conv on v; output VD layout [bh][e][n]
__global__ void vconv5(const float* __restrict__ kv, const float* __restrict__ dwc_w,
                       const float* __restrict__ dwc_b, float* __restrict__ VD) {
  int i = blockIdx.x * blockDim.x + threadIdx.x;
  const int total = BH * HDIM * SEQ_N;
  if (i >= total) return;
  int n = i % SEQ_N, e = (i / SEQ_N) % HDIM, bh = i / (SEQ_N * HDIM);
  int b = bh >> 3, h = bh & 7, c = h * HDIM + e;
  int y = n / IMG, x = n % IMG;
  float s = dwc_b[e];
  #pragma unroll
  for (int ky = 0; ky < 5; ++ky) {
    int yy = y + ky - 2;
    if (yy < 0 || yy >= IMG) continue;
    #pragma unroll
    for (int kx = 0; kx < 5; ++kx) {
      int xx = x + kx - 2;
      if (xx < 0 || xx >= IMG) continue;
      s += dwc_w[e * 25 + ky * 5 + kx] *
           kv[((long long)b * SEQ_N + yy * IMG + xx) * (2 * CH) + CH + c];
    }
  }
  VD[i] = s;
}

// out_pre = (xa*z + v_dwconv) * g  -> bf16 for projection GEMM
__global__ void combine(const float* __restrict__ XA, const float* __restrict__ VD,
                        const float* __restrict__ Z, const float* __restrict__ Z2,
                        const float* __restrict__ qg, __bf16* __restrict__ OB) {
  int i = blockIdx.x * blockDim.x + threadIdx.x;
  const int total = BATCH * SEQ_N * CH;
  if (i >= total) return;
  int c = i % CH, n = (i / CH) % SEQ_N, b = i / (CH * SEQ_N);
  int h = c >> 6, e = c & 63, bh = b * NHEADS + h;
  float z = (e < 32) ? Z[(long long)bh * SEQ_N + n] : Z2[(long long)bh * SEQ_N + n];
  float xa = XA[((long long)bh * SEQ_N + n) * HDIM + e] * z;
  float g  = qg[((long long)b * SEQ_N + n) * (2 * CH) + CH + c];
  float o  = (xa + VD[((long long)bh * HDIM + e) * SEQ_N + n]) * g;
  OB[i] = (__bf16)o;
}

// ---------------- host orchestration ----------------
extern "C" void kernel_launch(void* const* d_in, const int* in_sizes, int n_in,
                              void* d_out, int out_size, void* d_ws, size_t ws_size,
                              hipStream_t stream) {
  const float* x        = (const float*)d_in[0];
  const float* qg_w     = (const float*)d_in[1];
  const float* kv_w     = (const float*)d_in[2];
  const float* proj_w   = (const float*)d_in[3];
  const float* proj_b   = (const float*)d_in[4];
  const float* dwc_w    = (const float*)d_in[5];
  const float* dwc_b    = (const float*)d_in[6];
  const float* power_p  = (const float*)d_in[7];
  const float* scale_p  = (const float*)d_in[8];
  const float* enh_w    = (const float*)d_in[9];
  const float* bn_gamma = (const float*)d_in[10];
  const float* bn_beta  = (const float*)d_in[11];
  const float* enh_wt   = (const float*)d_in[12];
  const float* pos_enc  = (const float*)d_in[13];
  float* out = (float*)d_out;

  // workspace carve-up
  char* p = (char*)d_ws;
  auto alloc = [&](size_t bytes) { void* r = (void*)p; p += (bytes + 255) & ~(size_t)255; return r; };
  const long long MN = (long long)BATCH * SEQ_N;       // 25088
  __bf16* XB     = (__bf16*)alloc(MN * CH * 2);
  __bf16* QGWB   = (__bf16*)alloc((size_t)2 * CH * CH * 2);
  __bf16* KVWB   = (__bf16*)alloc((size_t)2 * CH * CH * 2);
  __bf16* PWB    = (__bf16*)alloc((size_t)CH * CH * 2);
  float*  QG     = (float*)alloc(MN * 2 * CH * 4);
  float*  KVf    = (float*)alloc(MN * 2 * CH * 4);
  float*  QE     = (float*)alloc(MN * CH * 4);
  float*  MEAN   = (float*)alloc(CH * 4);
  float*  RSTD   = (float*)alloc(CH * 4);
  __bf16* QS     = (__bf16*)alloc((size_t)BH * SEQ_N * 128 * 2);
  __bf16* QO     = (__bf16*)alloc((size_t)BH * SEQ_N * 128 * 2);
  __bf16* KFT    = (__bf16*)alloc((size_t)BH * 128 * SEQ_N * 2);
  __bf16* VB     = (__bf16*)alloc((size_t)BH * SEQ_N * HDIM * 2);
  float*  KMEAN  = (float*)alloc((size_t)BH * 128 * 4);
  float*  KVMAT  = (float*)alloc((size_t)BH * 128 * 64 * 4);
  __bf16* KVMATB = (__bf16*)alloc((size_t)BH * 128 * 64 * 2);
  float*  Z      = (float*)alloc((size_t)BH * SEQ_N * 4);
  float*  Z2     = (float*)alloc((size_t)BH * SEQ_N * 4);
  float*  XA     = (float*)alloc((size_t)BH * SEQ_N * HDIM * 4);
  float*  VD     = (float*)alloc((size_t)BH * HDIM * SEQ_N * 4);
  __bf16* OB     = (__bf16*)alloc(MN * CH * 2);

  auto g1 = [](int n) { return dim3((n + 255) / 256); };

  // bf16 casts
  cast_f32_bf16<<<g1((int)(MN * CH)), 256, 0, stream>>>(x, XB, (int)(MN * CH));
  cast_f32_bf16<<<g1(2 * CH * CH), 256, 0, stream>>>(qg_w, QGWB, 2 * CH * CH);
  cast_f32_bf16<<<g1(2 * CH * CH), 256, 0, stream>>>(kv_w, KVWB, 2 * CH * CH);
  cast_f32_bf16<<<g1(CH * CH), 256, 0, stream>>>(proj_w, PWB, CH * CH);

  // qg = x @ qg_w^T ; kv = x @ kv_w^T   (M=25088, Nout=1024, K=512)
  {
    dim3 grid((2 * CH) / BN, (int)((MN + BM - 1) / BM), 1);
    wmma_gemm_bf16<<<grid, 256, 0, stream>>>(XB, CH, 0, QGWB, CH, 0, QG, 2 * CH, 0,
                                             (int)MN, 2 * CH, CH, 1.0f, nullptr, 0);
    wmma_gemm_bf16<<<grid, 256, 0, stream>>>(XB, CH, 0, KVWB, CH, 0, KVf, 2 * CH, 0,
                                             (int)MN, 2 * CH, CH, 1.0f, nullptr, 0);
  }

  // enhance path: 3x3 dwconv -> BN stats -> fused q/k update
  enhance_conv3<<<g1((int)(MN * CH)), 256, 0, stream>>>(QG, enh_w, QE);
  bn_stats<<<dim3(CH), 256, 0, stream>>>(QE, MEAN, RSTD);
  fuse_qk<<<g1((int)(MN * CH)), 256, 0, stream>>>(QG, KVf, QE, MEAN, RSTD, bn_gamma,
                                                  bn_beta, scale_p, enh_wt, pos_enc);

  // polarity features + k_mean
  features<<<g1((int)(MN * CH)), 256, 0, stream>>>(QG, KVf, power_p, QS, QO, KFT, VB);
  kmean_kernel<<<dim3(BH * 128), 256, 0, stream>>>(KFT, KMEAN);

  // KV = (1/N) * kf^T @ v   per (b,h):  M=128, Nout=64, K=3136, B row-major [K,N]
  {
    dim3 grid(1, 1, BH);
    wmma_gemm_bf16<<<grid, 256, 0, stream>>>(
        KFT, SEQ_N, 128LL * SEQ_N, VB, HDIM, (long long)SEQ_N * HDIM,
        KVMAT, 64, 128LL * 64, 128, 64, SEQ_N, 1.0f / (float)SEQ_N, nullptr, 1);
  }
  cast_f32_bf16<<<g1(BH * 128 * 64), 256, 0, stream>>>(KVMAT, KVMATB, BH * 128 * 64);

  // x_sim = q_sim @ kv1 ; x_opp = q_opp @ kv2   (M=3136, Nout=32, K=128 each)
  {
    dim3 grid(1, (SEQ_N + BM - 1) / BM, BH);
    wmma_gemm_bf16<<<grid, 256, 0, stream>>>(
        QS, 128, (long long)SEQ_N * 128, KVMATB, 64, 128LL * 64,
        XA, 64, (long long)SEQ_N * 64, SEQ_N, 32, 128, 1.0f, nullptr, 1);
    wmma_gemm_bf16<<<grid, 256, 0, stream>>>(
        QO, 128, (long long)SEQ_N * 128, KVMATB + 32, 64, 128LL * 64,
        XA + 32, 64, (long long)SEQ_N * 64, SEQ_N, 32, 128, 1.0f, nullptr, 1);
  }

  // normalizers, 5x5 dwconv on v, gated combine
  zfactors<<<g1(BH * SEQ_N), 256, 0, stream>>>(QS, QO, KMEAN, Z, Z2);
  vconv5<<<g1(BH * HDIM * SEQ_N), 256, 0, stream>>>(KVf, dwc_w, dwc_b, VD);
  combine<<<g1((int)(MN * CH)), 256, 0, stream>>>(XA, VD, Z, Z2, QG, OB);

  // out = out_pre @ proj_w^T + proj_b   (M=25088, Nout=512, K=512)
  {
    dim3 grid(CH / BN, (int)((MN + BM - 1) / BM), 1);
    wmma_gemm_bf16<<<grid, 256, 0, stream>>>(OB, CH, 0, PWB, CH, 0, out, CH, 0,
                                             (int)MN, CH, CH, 1.0f, proj_b, 0);
  }
}
